// VirtualNodeNetwork_22917945491534
// MI455X (gfx1250) — compile-verified, matrix operand order checked by reading, not run on hardware
//
#include <hip/hip_runtime.h>
#include <math.h>

// ---------------------------------------------------------------------------
// CDNA5 / gfx1250 implementation of the e3nn-style VirtualNodeNetwork.
// Edge radial MLP runs on v_wmma_f32_16x16x32_f16 (16 edges per wave),
// fused with the CG tensor-product message + atomic segment-sum so the
// (E x 96) per-edge weight tensor is never materialized in HBM.
// Round-2 changes: fast rcp/rsq (no IEEE div chains), pre-swizzled LDS
// weight tiles -> single ds_load_b128 per WMMA operand, bank-conflict-free
// wbuf stride (97), vectorized global gathers.
// ---------------------------------------------------------------------------

typedef __attribute__((ext_vector_type(16))) _Float16 v16h;
typedef __attribute__((ext_vector_type(8)))  _Float16 v8h;
typedef __attribute__((ext_vector_type(8)))  float    v8f;

namespace {
constexpr int   TPB    = 256;
constexpr float INV2   = 0.70710678118f;   // 1/sqrt(2)
constexpr float INV8   = 0.35355339059f;   // 1/sqrt(8)
constexpr float INV16  = 0.25f;            // 1/sqrt(16)
constexpr float INV32  = 0.17677669529f;   // 1/sqrt(32)
constexpr float INV48  = 0.14433756729f;   // 1/sqrt(48)
constexpr float INVDEG = 0.25f;            // 1/sqrt(AVG_DEG=16)
constexpr float INVN   = 0.056523403f;     // 1/sqrt(AVG_NODES=313)
constexpr float CS_    = 0.38268343236f;   // sin(pi/8)
constexpr float CX_    = 0.92387953251f;   // cos(pi/8)
constexpr float SQ3    = 1.73205080757f;
constexpr float INVSQ3 = 0.57735026919f;
constexpr float PI_F   = 3.14159265359f;
constexpr float MAX_R  = 2.5f;
constexpr float RB_STEP = 2.5f / 7.0f;
// 1.14136 * e^2 * sqrt(8)  (radial-basis amplitude with sqrt(NB) folded in)
constexpr float RB_COEF = 23.8536f;
constexpr int   WROW   = 97;               // padded wbuf row (bank-conflict free)
}

static __device__ __forceinline__ float fast_rcp(float x) {
  return __builtin_amdgcn_rcpf(x);         // v_rcp_f32
}
static __device__ __forceinline__ float silu_f(float x) {
  return x * fast_rcp(1.0f + __expf(-x));
}

// ---------------------------------------------------------------------------
__global__ __launch_bounds__(TPB) void zero_kernel(float* __restrict__ p, int n) {
  int i = blockIdx.x * TPB + threadIdx.x;
  if (i < n) p[i] = 0.0f;
}

// ---------------------------------------------------------------------------
// Edge geometry: y0 (cutoff), y1 (l=1 edge attr), emb (8 radial basis values)
__global__ __launch_bounds__(TPB) void edge_geom_kernel(
    const float* __restrict__ pos, const int* __restrict__ esrc,
    const int* __restrict__ edst, float* __restrict__ y0,
    float* __restrict__ y1, float* __restrict__ emb, int n_edges) {
  int e = blockIdx.x * TPB + threadIdx.x;
  if (e >= n_edges) return;
  int s = esrc[e], d = edst[e];
  float vx = pos[d * 3 + 0] - pos[s * 3 + 0];
  float vy = pos[d * 3 + 1] - pos[s * 3 + 1];
  float vz = pos[d * 3 + 2] - pos[s * 3 + 2];
  float r2 = vx * vx + vy * vy + vz * vz + 1e-18f;
  float invr = __builtin_amdgcn_rsqf(r2);  // v_rsq_f32: 1/sqrt(r2)
  float r = r2 * invr;
  float u = 2.0f * (r * (1.0f / MAX_R) - 1.0f);
  float cut;
  if (u > 0.0f)       cut = 0.0f;
  else if (u < -1.0f) cut = 1.0f;
  else                cut = 0.5f * (1.0f - __cosf(PI_F * u));
  y0[e] = cut;
  float sc = SQ3 * invr * cut;
  y1[e * 3 + 0] = sc * vx;
  y1[e * 3 + 1] = sc * vy;
  y1[e * 3 + 2] = sc * vz;
#pragma unroll
  for (int k = 0; k < 8; ++k) {
    float dd = (r - RB_STEP * (float)k) * (1.0f / RB_STEP);
    float val = 0.0f;
    if (fabsf(dd) < 1.0f) {
      float z = dd * dd - 1.0f;            // in [-1, 0)
      val = RB_COEF * __expf(fast_rcp(z));
    }
    emb[e * 8 + k] = val;
  }
}

// ---------------------------------------------------------------------------
// lin1: s1 = (s @ l1_s)/sqrt(32)*a ; v1 = channel-mix(v, l1_v)/sqrt(16)*a
__global__ __launch_bounds__(TPB) void lin1_kernel(
    const float* __restrict__ s_in, const float* __restrict__ v_in,
    const float* __restrict__ attr, const float* __restrict__ l1_s,
    const float* __restrict__ l1_v, float* __restrict__ s1,
    float* __restrict__ v1, int n_nodes) {
  __shared__ float Ws[32 * 32];
  __shared__ float Wv[16 * 16];
  for (int i = threadIdx.x; i < 1024; i += TPB) Ws[i] = l1_s[i] * INV32;
  for (int i = threadIdx.x; i < 256;  i += TPB) Wv[i] = l1_v[i] * INV16;
  __syncthreads();
  int n = blockIdx.x * TPB + threadIdx.x;
  if (n >= n_nodes) return;
  float a = attr[n];
  float sreg[32];
#pragma unroll
  for (int c4 = 0; c4 < 8; ++c4) {
    float4 t = ((const float4*)(s_in + (size_t)n * 32))[c4];
    sreg[c4 * 4 + 0] = t.x; sreg[c4 * 4 + 1] = t.y;
    sreg[c4 * 4 + 2] = t.z; sreg[c4 * 4 + 3] = t.w;
  }
  for (int o = 0; o < 32; ++o) {
    float acc = 0.0f;
#pragma unroll
    for (int c = 0; c < 32; ++c) acc += sreg[c] * Ws[c * 32 + o];
    s1[(size_t)n * 32 + o] = a * acc;
  }
  float vreg[48];
#pragma unroll
  for (int k4 = 0; k4 < 12; ++k4) {
    float4 t = ((const float4*)(v_in + (size_t)n * 48))[k4];
    vreg[k4 * 4 + 0] = t.x; vreg[k4 * 4 + 1] = t.y;
    vreg[k4 * 4 + 2] = t.z; vreg[k4 * 4 + 3] = t.w;
  }
  for (int o = 0; o < 16; ++o) {
    float ax = 0.0f, ay = 0.0f, az = 0.0f;
#pragma unroll
    for (int c = 0; c < 16; ++c) {
      float w = Wv[c * 16 + o];
      ax += vreg[c * 3 + 0] * w;
      ay += vreg[c * 3 + 1] * w;
      az += vreg[c * 3 + 2] * w;
    }
    v1[((size_t)n * 16 + o) * 3 + 0] = a * ax;
    v1[((size_t)n * 16 + o) * 3 + 1] = a * ay;
    v1[((size_t)n * 16 + o) * 3 + 2] = a * az;
  }
}

// ---------------------------------------------------------------------------
// Fused edge kernel: per-wave 16 edges.
//   GEMM1 (WMMA f16): h = silu(emb @ W1/sqrt8)     [16x8 -> 16x16]
//   GEMM2 (WMMA f16): w = h @ W2/sqrt16            [16x16 -> 16x96, 6 tiles]
//   then CG tensor-product messages + atomic segment-sum into ss/sv.
__global__ __launch_bounds__(TPB) void edge_msg_kernel(
    const float* __restrict__ emb, const float* __restrict__ y0,
    const float* __restrict__ y1, const int* __restrict__ esrc,
    const int* __restrict__ edst, const float* __restrict__ s1,
    const float* __restrict__ v1, const float* __restrict__ fc_w1,
    const float* __restrict__ fc_w2, float* __restrict__ ss,
    float* __restrict__ sv, int n_edges) {
  // Pre-swizzled weights: each WMMA B-operand pack is one 16B ds_load_b128.
  __shared__ _Float16 sW1p[16 * 8];          // [n][j]        = W1[j][n]*inv8
  __shared__ _Float16 sW2p[6 * 16 * 2 * 8];  // [t][n][hi][j] = W2[hi*8+j][16t+n]*inv16
  __shared__ _Float16 hbuf[8][16 * 16];      // per-wave h tile [edge][hidden]
  __shared__ float    wbuf[8][16 * WROW];    // per-wave w tile [edge][feature], padded

  const int tid = threadIdx.x;
  for (int i = tid; i < 128; i += TPB) {
    int n = i >> 3, j = i & 7;
    sW1p[i] = (_Float16)(fc_w1[j * 16 + n] * INV8);
  }
  for (int i = tid; i < 1536; i += TPB) {
    int j = i & 7, h2 = (i >> 3) & 1, n = (i >> 4) & 15, t = i >> 8;
    sW2p[i] = (_Float16)(fc_w2[(h2 * 8 + j) * 96 + t * 16 + n] * INV16);
  }
  __syncthreads();

  const int wave = tid >> 5;
  const int lane = tid & 31;
  const int m    = lane & 15;   // row within 16x16 tile (non-K dim)
  const int hi   = lane >> 4;   // half-wave K-slot selector
  const int ebase = blockIdx.x * 128 + wave * 16;

  const v8h z8 = {};

  // ---- GEMM1: A = emb rows (16 edges x K<=8, zero padded to K=32) ----
  v16h a = {};
  v8h bv = z8;
  if (hi == 0) {
    int et = ebase + m;
    if (et >= n_edges) et = n_edges - 1;
    const float4* er = (const float4*)(emb + (size_t)et * 8);
    float4 e0 = er[0], e1 = er[1];
    a[0] = (_Float16)e0.x; a[1] = (_Float16)e0.y;
    a[2] = (_Float16)e0.z; a[3] = (_Float16)e0.w;
    a[4] = (_Float16)e1.x; a[5] = (_Float16)e1.y;
    a[6] = (_Float16)e1.z; a[7] = (_Float16)e1.w;
    bv = *(const v8h*)(&sW1p[m * 8]);
  }
  v16h b = __builtin_shufflevector(bv, z8, 0, 1, 2, 3, 4, 5, 6, 7,
                                   8, 9, 10, 11, 12, 13, 14, 15);
  v8f c0 = {};
  v8f h = __builtin_amdgcn_wmma_f32_16x16x32_f16(
      false, a, false, b, (short)0, c0, false, false);

  // silu + stage h tile to LDS for the edge<->K transpose
  _Float16* hb = hbuf[wave];
#pragma unroll
  for (int r = 0; r < 8; ++r)                // C layout: row = r+8*hi, col = m
    hb[(r + 8 * hi) * 16 + m] = (_Float16)silu_f(h[r]);
  asm volatile("s_wait_dscnt 0" ::: "memory");  // same-wave DS in-order barrier

  // ---- GEMM2: A2 = h (16 edges x K=16, padded to 32) ----
  v8h av = *(const v8h*)(&hb[m * 16 + hi * 8]);   // one ds_load_b128
  v16h a2 = __builtin_shufflevector(av, z8, 0, 1, 2, 3, 4, 5, 6, 7,
                                    8, 9, 10, 11, 12, 13, 14, 15);

  float* wb = wbuf[wave];
#pragma unroll
  for (int t = 0; t < 6; ++t) {
    v8h bv2 = *(const v8h*)(&sW2p[((t * 16 + m) * 2 + hi) * 8]);  // ds_load_b128
    v16h b2 = __builtin_shufflevector(bv2, z8, 0, 1, 2, 3, 4, 5, 6, 7,
                                      8, 9, 10, 11, 12, 13, 14, 15);
    v8f cz = {};
    v8f wo = __builtin_amdgcn_wmma_f32_16x16x32_f16(
        false, a2, false, b2, (short)0, cz, false, false);
#pragma unroll
    for (int r = 0; r < 8; ++r)
      wb[(r + 8 * hi) * WROW + t * 16 + m] = wo[r];
  }
  asm volatile("s_wait_dscnt 0" ::: "memory");

  // ---- message phase: 2 lanes per edge, split 48 output channels ----
  const int k    = lane >> 1;
  const int half = lane & 1;
  const int e = ebase + k;
  if (e < n_edges) {
    const int src = esrc[e], dst = edst[e];
    const float cy0 = y0[e];
    const float y1x = y1[e * 3 + 0], y1y = y1[e * 3 + 1], y1z = y1[e * 3 + 2];
    const float* we  = wb + k * WROW;
    const float* se  = s1 + (size_t)src * 32;
    const float* ve  = v1 + (size_t)src * 48;
    float* ssd = ss + (size_t)dst * 48;
    float* svd = sv + (size_t)dst * 144;

    const int c_lo = half ? 24 : 0;
    const int c_n4 = half ? 2 : 6;          // float4 groups of scalar-source chans
    for (int g = 0; g < c_n4; ++g) {
      int c = c_lo + g * 4;
      float4 s4 = *(const float4*)(se + c);  // global_load_b128 gather
      float sc4[4] = {s4.x, s4.y, s4.z, s4.w};
#pragma unroll
      for (int q = 0; q < 4; ++q) {
        int cc = c + q;
        float sc = sc4[q];
        atomicAdd(&ssd[cc], INVDEG * we[cc] * sc * cy0);        // 0e x 0e -> 0e
        float w2c = INVDEG * we[32 + cc] * sc;                   // 0e x 1o -> 1o
        atomicAdd(&svd[cc * 3 + 0], w2c * y1x);
        atomicAdd(&svd[cc * 3 + 1], w2c * y1y);
        atomicAdd(&svd[cc * 3 + 2], w2c * y1z);
      }
    }
    if (half) {                              // vector-source paths
      float vr[48];
#pragma unroll
      for (int g = 0; g < 12; ++g) {
        float4 t4 = ((const float4*)ve)[g];
        vr[g * 4 + 0] = t4.x; vr[g * 4 + 1] = t4.y;
        vr[g * 4 + 2] = t4.z; vr[g * 4 + 3] = t4.w;
      }
      for (int c1 = 0; c1 < 16; ++c1) {
        float vx = vr[c1 * 3 + 0], vy = vr[c1 * 3 + 1], vz = vr[c1 * 3 + 2];
        float dot = (vx * y1x + vy * y1y + vz * y1z) * INVSQ3;  // 1o x 1o -> 0e
        atomicAdd(&ssd[32 + c1], INVDEG * we[80 + c1] * dot);
        float w3 = INVDEG * we[64 + c1] * cy0;                   // 1o x 0e -> 1o
        atomicAdd(&svd[(32 + c1) * 3 + 0], w3 * vx);
        atomicAdd(&svd[(32 + c1) * 3 + 1], w3 * vy);
        atomicAdd(&svd[(32 + c1) * 3 + 2], w3 * vz);
      }
    }
  }
}

// ---------------------------------------------------------------------------
// conv combine: out = CS*sc(x) + CX*lin2(segment_sum)
__global__ __launch_bounds__(TPB) void conv_combine_kernel(
    const float* __restrict__ s_in, const float* __restrict__ v_in,
    const float* __restrict__ attr, const float* __restrict__ ss,
    const float* __restrict__ sv, const float* __restrict__ sc_s,
    const float* __restrict__ sc_v, const float* __restrict__ l2_s,
    const float* __restrict__ l2_v, float* __restrict__ s_out,
    float* __restrict__ v_out, int n_nodes) {
  __shared__ float WscS[32 * 32], Wl2S[48 * 32], WscV[16 * 16], Wl2V[48 * 16];
  for (int i = threadIdx.x; i < 1024; i += TPB) WscS[i] = sc_s[i] * (CS_ * INV32);
  for (int i = threadIdx.x; i < 1536; i += TPB) Wl2S[i] = l2_s[i] * (CX_ * INV48);
  for (int i = threadIdx.x; i < 256;  i += TPB) WscV[i] = sc_v[i] * (CS_ * INV16);
  for (int i = threadIdx.x; i < 768;  i += TPB) Wl2V[i] = l2_v[i] * (CX_ * INV48);
  __syncthreads();
  int n = blockIdx.x * TPB + threadIdx.x;
  if (n >= n_nodes) return;
  float aat = attr[n];

  float accs[32];
#pragma unroll
  for (int o = 0; o < 32; ++o) accs[o] = 0.0f;
  for (int c = 0; c < 32; ++c) {
    float x = s_in[(size_t)n * 32 + c];
#pragma unroll
    for (int o = 0; o < 32; ++o) accs[o] += x * WscS[c * 32 + o];
  }
  for (int c = 0; c < 48; ++c) {
    float x = ss[(size_t)n * 48 + c];
#pragma unroll
    for (int o = 0; o < 32; ++o) accs[o] += x * Wl2S[c * 32 + o];
  }
#pragma unroll
  for (int o = 0; o < 32; ++o) s_out[(size_t)n * 32 + o] = aat * accs[o];

  float accv[48];
#pragma unroll
  for (int k = 0; k < 48; ++k) accv[k] = 0.0f;
  for (int c = 0; c < 16; ++c) {
    float x = v_in[((size_t)n * 16 + c) * 3 + 0];
    float y = v_in[((size_t)n * 16 + c) * 3 + 1];
    float z = v_in[((size_t)n * 16 + c) * 3 + 2];
#pragma unroll
    for (int o = 0; o < 16; ++o) {
      float w = WscV[c * 16 + o];
      accv[o * 3 + 0] += x * w; accv[o * 3 + 1] += y * w; accv[o * 3 + 2] += z * w;
    }
  }
  for (int c = 0; c < 48; ++c) {
    float x = sv[(size_t)n * 144 + c * 3 + 0];
    float y = sv[(size_t)n * 144 + c * 3 + 1];
    float z = sv[(size_t)n * 144 + c * 3 + 2];
#pragma unroll
    for (int o = 0; o < 16; ++o) {
      float w = Wl2V[c * 16 + o];
      accv[o * 3 + 0] += x * w; accv[o * 3 + 1] += y * w; accv[o * 3 + 2] += z * w;
    }
  }
#pragma unroll
  for (int k = 0; k < 48; ++k) v_out[(size_t)n * 48 + k] = aat * accv[k];
}

// ---------------------------------------------------------------------------
// VN node->virtual messages (atomic segment sum over batch)
__global__ __launch_bounds__(TPB) void vn_msg_kernel(
    const float* __restrict__ s_in, const float* __restrict__ v_in,
    const float* __restrict__ pos, const int* __restrict__ batch,
    const float* __restrict__ tp_ss, const float* __restrict__ tp_vs,
    const float* __restrict__ tp_sv, const float* __restrict__ tp_vv,
    float* __restrict__ gms, float* __restrict__ gmv, int n_nodes) {
  __shared__ float Wss[32 * 32], Wvs[16 * 32], Wsv[32 * 16], Wvv[16 * 16];
  for (int i = threadIdx.x; i < 1024; i += TPB) Wss[i] = tp_ss[i] * (INV32 * INV2);
  for (int i = threadIdx.x; i < 512;  i += TPB) Wvs[i] = tp_vs[i] * (INV16 * INV2);
  for (int i = threadIdx.x; i < 512;  i += TPB) Wsv[i] = tp_sv[i] * (INV32 * INV2);
  for (int i = threadIdx.x; i < 256;  i += TPB) Wvv[i] = tp_vv[i] * (INV16 * INV2);
  __syncthreads();
  int n = blockIdx.x * TPB + threadIdx.x;
  if (n >= n_nodes) return;
  int b = batch[n];
  const float p1x = pos[n * 3 + 0] * (SQ3 / 5.0f);
  const float p1y = pos[n * 3 + 1] * (SQ3 / 5.0f);
  const float p1z = pos[n * 3 + 2] * (SQ3 / 5.0f);
  float sreg[32], vreg[48];
#pragma unroll
  for (int c = 0; c < 32; ++c) sreg[c] = s_in[(size_t)n * 32 + c];
#pragma unroll
  for (int k = 0; k < 48; ++k) vreg[k] = v_in[(size_t)n * 48 + k];

  float ms[32];
#pragma unroll
  for (int o = 0; o < 32; ++o) ms[o] = 0.0f;
  for (int c = 0; c < 32; ++c) {
    float x = sreg[c];
#pragma unroll
    for (int o = 0; o < 32; ++o) ms[o] += x * Wss[c * 32 + o];
  }
  for (int c1 = 0; c1 < 16; ++c1) {
    float dot = (vreg[c1 * 3 + 0] * p1x + vreg[c1 * 3 + 1] * p1y +
                 vreg[c1 * 3 + 2] * p1z) * INVSQ3;
#pragma unroll
    for (int o = 0; o < 32; ++o) ms[o] += dot * Wvs[c1 * 32 + o];
  }
#pragma unroll
  for (int o = 0; o < 32; ++o) atomicAdd(&gms[b * 32 + o], ms[o]);

  for (int o = 0; o < 16; ++o) {
    float t = 0.0f;
#pragma unroll
    for (int c = 0; c < 32; ++c) t += sreg[c] * Wsv[c * 16 + o];
    float mx = t * p1x, my = t * p1y, mz = t * p1z;
#pragma unroll
    for (int c1 = 0; c1 < 16; ++c1) {
      float w = Wvv[c1 * 16 + o];
      mx += vreg[c1 * 3 + 0] * w; my += vreg[c1 * 3 + 1] * w; mz += vreg[c1 * 3 + 2] * w;
    }
    atomicAdd(&gmv[(b * 16 + o) * 3 + 0], mx);
    atomicAdd(&gmv[(b * 16 + o) * 3 + 1], my);
    atomicAdd(&gmv[(b * 16 + o) * 3 + 2], mz);
  }
}

// ---------------------------------------------------------------------------
// VN update (single block): vs_new, vv_new and the node-broadcast features
__global__ __launch_bounds__(TPB) void vn_update_kernel(
    const float* __restrict__ vs_in, const float* __restrict__ vv_in,
    const float* __restrict__ gms, const float* __restrict__ gmv,
    const float* __restrict__ vsc_s, const float* __restrict__ vsc_v,
    const float* __restrict__ n2v_s, const float* __restrict__ n2v_v,
    float* __restrict__ vs_out, float* __restrict__ vv_out,
    float* __restrict__ nsg, float* __restrict__ nvg) {
  __shared__ float Lvs[64 * 32];
  __shared__ float Lvv[64 * 48];
  for (int idx = threadIdx.x; idx < 64 * 32; idx += TPB) {
    int g = idx >> 5, o = idx & 31;
    float acc = 0.0f;
    for (int c = 0; c < 32; ++c) acc += vs_in[g * 32 + c] * vsc_s[c * 32 + o];
    float m = silu_f(gms[idx] * INVN);
    float val = (acc * INV32 + m) * INV2;
    Lvs[idx] = val; vs_out[idx] = val;
  }
  for (int idx = threadIdx.x; idx < 64 * 48; idx += TPB) {
    int g = idx / 48, rem = idx % 48, o = rem / 3, i3 = rem % 3;
    float acc = 0.0f;
    for (int c = 0; c < 16; ++c) acc += vv_in[(g * 16 + c) * 3 + i3] * vsc_v[c * 16 + o];
    float val = (acc * INV16 + gmv[idx] * INVN) * INV2;
    Lvv[idx] = val; vv_out[idx] = val;
  }
  __syncthreads();
  for (int idx = threadIdx.x; idx < 64 * 32; idx += TPB) {
    int g = idx >> 5, o = idx & 31;
    float acc = 0.0f;
    for (int c = 0; c < 32; ++c) acc += Lvs[g * 32 + c] * n2v_s[c * 32 + o];
    nsg[idx] = silu_f(acc * INV32);
  }
  for (int idx = threadIdx.x; idx < 64 * 48; idx += TPB) {
    int g = idx / 48, rem = idx % 48, o = rem / 3, i3 = rem % 3;
    float acc = 0.0f;
    for (int c = 0; c < 16; ++c) acc += Lvv[(g * 16 + c) * 3 + i3] * n2v_v[c * 16 + o];
    nvg[idx] = acc * INV16;
  }
}

// ---------------------------------------------------------------------------
// VN virtual->node scatter + self-connection
__global__ __launch_bounds__(TPB) void vn_scatter_kernel(
    const float* __restrict__ s_in, const float* __restrict__ v_in,
    const int* __restrict__ batch, const float* __restrict__ nsg,
    const float* __restrict__ nvg, const float* __restrict__ nsc_s,
    const float* __restrict__ nsc_v, float* __restrict__ s_out,
    float* __restrict__ v_out, int n_nodes) {
  __shared__ float Ws[32 * 32], Wv[16 * 16];
  for (int i = threadIdx.x; i < 1024; i += TPB) Ws[i] = nsc_s[i] * INV32;
  for (int i = threadIdx.x; i < 256;  i += TPB) Wv[i] = nsc_v[i] * INV16;
  __syncthreads();
  int n = blockIdx.x * TPB + threadIdx.x;
  if (n >= n_nodes) return;
  int b = batch[n];
  float sreg[32], vreg[48];
#pragma unroll
  for (int c = 0; c < 32; ++c) sreg[c] = s_in[(size_t)n * 32 + c];
#pragma unroll
  for (int k = 0; k < 48; ++k) vreg[k] = v_in[(size_t)n * 48 + k];
  for (int o = 0; o < 32; ++o) {
    float acc = 0.0f;
#pragma unroll
    for (int c = 0; c < 32; ++c) acc += sreg[c] * Ws[c * 32 + o];
    s_out[(size_t)n * 32 + o] = (acc + nsg[b * 32 + o]) * INV2;
  }
  for (int o = 0; o < 16; ++o) {
    float ax = 0.0f, ay = 0.0f, az = 0.0f;
#pragma unroll
    for (int c = 0; c < 16; ++c) {
      float w = Wv[c * 16 + o];
      ax += vreg[c * 3 + 0] * w; ay += vreg[c * 3 + 1] * w; az += vreg[c * 3 + 2] * w;
    }
    v_out[((size_t)n * 16 + o) * 3 + 0] = (ax + nvg[(b * 16 + o) * 3 + 0]) * INV2;
    v_out[((size_t)n * 16 + o) * 3 + 1] = (ay + nvg[(b * 16 + o) * 3 + 1]) * INV2;
    v_out[((size_t)n * 16 + o) * 3 + 2] = (az + nvg[(b * 16 + o) * 3 + 2]) * INV2;
  }
}

// ---------------------------------------------------------------------------
__global__ __launch_bounds__(TPB) void out_kernel(
    const float* __restrict__ s_in, const float* __restrict__ out_w,
    float* __restrict__ out, int n_nodes) {
  int n = blockIdx.x * TPB + threadIdx.x;
  if (n >= n_nodes) return;
  float o0 = 0.0f, o1 = 0.0f, o2 = 0.0f;
#pragma unroll
  for (int c = 0; c < 32; ++c) {
    float x = s_in[(size_t)n * 32 + c];
    o0 += x * out_w[c * 3 + 0];
    o1 += x * out_w[c * 3 + 1];
    o2 += x * out_w[c * 3 + 2];
  }
  out[n * 3 + 0] = o0 * INV32;
  out[n * 3 + 1] = o1 * INV32;
  out[n * 3 + 2] = o2 * INV32;
}

// ---------------------------------------------------------------------------
extern "C" void kernel_launch(void* const* d_in, const int* in_sizes, int n_in,
                              void* d_out, int out_size, void* d_ws, size_t ws_size,
                              hipStream_t stream) {
  (void)out_size; (void)ws_size; (void)n_in;
  // ---- input map: depth-first insertion order of setup_inputs() dict ----
  const float* pos   = (const float*)d_in[0];
  const float* s0    = (const float*)d_in[1];
  const float* v0    = (const float*)d_in[2];
  const float* vs0   = (const float*)d_in[3];
  const float* vv0   = (const float*)d_in[4];
  const float* attr  = (const float*)d_in[5];
  const float* out_w = (const float*)d_in[6];
  struct ConvP { const float *sc_s, *sc_v, *l1_s, *l1_v, *fc_w1, *fc_w2, *l2_s, *l2_v; };
  struct VnP   { const float *vsc_s, *vsc_v, *nsc_s, *nsc_v, *tp_ss, *tp_vs,
                 *tp_sv, *tp_vv, *n2v_s, *n2v_v; };
  ConvP conv[2]; VnP vn[2];
  int idx = 7;
  for (int L = 0; L < 2; ++L) {
    conv[L].sc_s  = (const float*)d_in[idx + 0];
    conv[L].sc_v  = (const float*)d_in[idx + 1];
    conv[L].l1_s  = (const float*)d_in[idx + 2];
    conv[L].l1_v  = (const float*)d_in[idx + 3];
    conv[L].fc_w1 = (const float*)d_in[idx + 4];
    conv[L].fc_w2 = (const float*)d_in[idx + 5];
    conv[L].l2_s  = (const float*)d_in[idx + 6];
    conv[L].l2_v  = (const float*)d_in[idx + 7];
    idx += 8;
    vn[L].vsc_s = (const float*)d_in[idx + 0];
    vn[L].vsc_v = (const float*)d_in[idx + 1];
    vn[L].nsc_s = (const float*)d_in[idx + 2];
    vn[L].nsc_v = (const float*)d_in[idx + 3];
    vn[L].tp_ss = (const float*)d_in[idx + 4];
    vn[L].tp_vs = (const float*)d_in[idx + 5];
    vn[L].tp_sv = (const float*)d_in[idx + 6];
    vn[L].tp_vv = (const float*)d_in[idx + 7];
    vn[L].n2v_s = (const float*)d_in[idx + 8];
    vn[L].n2v_v = (const float*)d_in[idx + 9];
    idx += 10;
  }
  const int* esrc  = (const int*)d_in[idx + 0];
  const int* edst  = (const int*)d_in[idx + 1];
  const int* batch = (const int*)d_in[idx + 2];
  const int N = in_sizes[1] / 32;
  const int E = in_sizes[idx + 0];
  const int G = in_sizes[3] / 32;

  // ---- workspace carve ----
  float* w = (float*)d_ws;
  auto carve = [&](size_t nelem) { float* p = w; w += nelem; return p; };
  float* y0  = carve((size_t)E);
  float* y1  = carve((size_t)E * 3);
  float* emb = carve((size_t)E * 8);
  float* sA  = carve((size_t)N * 32);
  float* vA  = carve((size_t)N * 48);
  float* sB  = carve((size_t)N * 32);
  float* vB  = carve((size_t)N * 48);
  float* s1  = carve((size_t)N * 32);
  float* v1  = carve((size_t)N * 48);
  float* ss  = carve((size_t)N * 48);
  float* sv  = carve((size_t)N * 144);   // contiguous with ss: zeroed together
  float* vsA = carve((size_t)G * 32);
  float* vvA = carve((size_t)G * 48);
  float* vsB = carve((size_t)G * 32);
  float* vvB = carve((size_t)G * 48);
  float* gms = carve((size_t)G * 32);
  float* gmv = carve((size_t)G * 48);    // contiguous with gms
  float* nsg = carve((size_t)G * 32);
  float* nvg = carve((size_t)G * 48);

  auto nb = [](int x) { return (x + TPB - 1) / TPB; };
  const int eb = (E + 127) / 128;        // 8 waves x 16 edges per block

  edge_geom_kernel<<<nb(E), TPB, 0, stream>>>(pos, esrc, edst, y0, y1, emb, E);

  const float* curS = s0;  const float* curV = v0;
  const float* curVS = vs0; const float* curVV = vv0;
  for (int L = 0; L < 2; ++L) {
    float* vsOut = (L == 0) ? vsA : vsB;
    float* vvOut = (L == 0) ? vvA : vvB;
    lin1_kernel<<<nb(N), TPB, 0, stream>>>(curS, curV, attr, conv[L].l1_s,
                                           conv[L].l1_v, s1, v1, N);
    zero_kernel<<<nb(N * 192), TPB, 0, stream>>>(ss, N * 192);
    edge_msg_kernel<<<eb, TPB, 0, stream>>>(emb, y0, y1, esrc, edst, s1, v1,
                                            conv[L].fc_w1, conv[L].fc_w2, ss, sv, E);
    conv_combine_kernel<<<nb(N), TPB, 0, stream>>>(curS, curV, attr, ss, sv,
        conv[L].sc_s, conv[L].sc_v, conv[L].l2_s, conv[L].l2_v, sB, vB, N);
    zero_kernel<<<nb(G * 80), TPB, 0, stream>>>(gms, G * 80);
    vn_msg_kernel<<<nb(N), TPB, 0, stream>>>(sB, vB, pos, batch,
        vn[L].tp_ss, vn[L].tp_vs, vn[L].tp_sv, vn[L].tp_vv, gms, gmv, N);
    vn_update_kernel<<<1, TPB, 0, stream>>>(curVS, curVV, gms, gmv,
        vn[L].vsc_s, vn[L].vsc_v, vn[L].n2v_s, vn[L].n2v_v,
        vsOut, vvOut, nsg, nvg);
    vn_scatter_kernel<<<nb(N), TPB, 0, stream>>>(sB, vB, batch, nsg, nvg,
        vn[L].nsc_s, vn[L].nsc_v, sA, vA, N);
    curS = sA; curV = vA; curVS = vsOut; curVV = vvOut;
  }
  out_kernel<<<nb(N), TPB, 0, stream>>>(sA, out_w, (float*)d_out, N);
}
